// CommunicatingAgentSystem_32031866094040
// MI455X (gfx1250) — matrix operand order repaired and checked by reading, not compiled, not run on hardware
//
#include <hip/hip_runtime.h>
#include <hip/hip_bf16.h>
#include <math.h>

typedef __attribute__((ext_vector_type(16))) _Float16 v16h;
typedef __attribute__((ext_vector_type(8)))  float    v8f;

#define AGENTS 5
#define EDIM   256
#define DDIM   512
#define NSUP   2048
#define NQRY   8192
#define NWAYC  5

#define TILE_M 64
#define TILE_N 64
#define TILE_K 32

__device__ __forceinline__ float apply_act(float v, int act) {
    switch (act) {
        case 1:  return fmaxf(v, 0.0f);
        case 2:  return tanhf(v);
        case 3:  return 1.0f / (1.0f + expf(-v));
        default: return v;
    }
}

// ---------------------------------------------------------------------------
// Generic WMMA GEMM: C[M,N] = act(alpha * (A[M,K] @ B[K,N]) + bias[N])
// A: f32 row-major, lda.  B: f32, ldb; transB=1 means B[k][n] = Bp[n*ldb+k].
// Block = 256 threads = 8 waves; block tile 64x64; each wave: two 16x16 tiles.
// A/B converted f32->f16 while staging into LDS; accumulate in f32 via
// v_wmma_f32_16x16x32_f16.
// ---------------------------------------------------------------------------
__global__ __launch_bounds__(256)
void wmma_gemm_kernel(const float* __restrict__ A, int lda,
                      const float* __restrict__ B, int ldb, int transB,
                      float* __restrict__ C, int ldc,
                      int M, int N, int K,
                      const float* __restrict__ bias,
                      float alpha, int act)
{
    __shared__ _Float16 sA[TILE_M][TILE_K];   // 64x32 f16 = 4 KB
    __shared__ _Float16 sB[TILE_K][TILE_N];   // 32x64 f16 = 4 KB

    const int tid  = threadIdx.x;
    const int lane = tid & 31;
    const int wave = tid >> 5;                // 0..7
    const int m0 = blockIdx.y * TILE_M;
    const int n0 = blockIdx.x * TILE_N;

    const int mt  = wave >> 1;                // 16-row tile index (0..3)
    const int ntb = (wave & 1) * 2;           // first of two 16-col tiles

    v8f acc0 = {}; v8f acc1 = {};

    const int frag_mn = lane & 15;            // row (A) / col (B,C) within tile
    const int hiA = (lane >> 4) ? 8  : 0;     // A K-half offset per ISA table
    const int hiB = (lane >> 4) ? 8  : 0;     // B K-half offset per ISA table

    for (int k0 = 0; k0 < K; k0 += TILE_K) {
        // stage A tile (2048 elems / 256 threads)
        for (int i = tid; i < TILE_M * TILE_K; i += 256) {
            int r = i >> 5, c = i & 31;
            sA[r][c] = (_Float16)A[(size_t)(m0 + r) * lda + (k0 + c)];
        }
        // stage B tile
        for (int i = tid; i < TILE_K * TILE_N; i += 256) {
            int r = i >> 6, c = i & 63;
            float v = transB ? B[(size_t)(n0 + c) * ldb + (k0 + r)]
                             : B[(size_t)(k0 + r) * ldb + (n0 + c)];
            sB[r][c] = (_Float16)v;
        }
        __syncthreads();

        v16h a;
        #pragma unroll
        for (int r = 0; r < 8; ++r) {
            int k = ((r < 4) ? (2 * r) : (16 + 2 * (r - 4))) + hiA;
            a[2 * r]     = sA[mt * 16 + frag_mn][k];
            a[2 * r + 1] = sA[mt * 16 + frag_mn][k + 1];
        }
        v16h b0, b1;
        #pragma unroll
        for (int r = 0; r < 8; ++r) {
            int k = ((r < 4) ? (2 * r) : (16 + 2 * (r - 4))) + hiB;
            b0[2 * r]     = sB[k][ntb * 16 + frag_mn];
            b0[2 * r + 1] = sB[k + 1][ntb * 16 + frag_mn];
            b1[2 * r]     = sB[k][(ntb + 1) * 16 + frag_mn];
            b1[2 * r + 1] = sB[k + 1][(ntb + 1) * 16 + frag_mn];
        }
        acc0 = __builtin_amdgcn_wmma_f32_16x16x32_f16(false, a, false, b0,
                                                      (short)0, acc0, false, false);
        acc1 = __builtin_amdgcn_wmma_f32_16x16x32_f16(false, a, false, b1,
                                                      (short)0, acc1, false, false);
        __syncthreads();
    }

    const int rowHi = (lane >> 4) ? 8 : 0;    // C/D layout: VGPR r -> row r(+8)
    #pragma unroll
    for (int r = 0; r < 8; ++r) {
        int m = m0 + mt * 16 + r + rowHi;
        int n1 = n0 + ntb * 16 + frag_mn;
        int n2 = n0 + (ntb + 1) * 16 + frag_mn;
        float v1 = alpha * acc0[r] + (bias ? bias[n1] : 0.0f);
        float v2 = alpha * acc1[r] + (bias ? bias[n2] : 0.0f);
        C[(size_t)m * ldc + n1] = apply_act(v1, act);
        C[(size_t)m * ldc + n2] = apply_act(v2, act);
    }
}

// ---------------------------------------------------------------------------
// Row LayerNorm + activation, in place. One block per row.
// ---------------------------------------------------------------------------
__global__ __launch_bounds__(256)
void ln_act_kernel(float* __restrict__ x, int n,
                   const float* __restrict__ g, const float* __restrict__ b, int act)
{
    __shared__ float red[256];
    float* row = x + (size_t)blockIdx.x * n;
    int t = threadIdx.x;

    float s = 0.0f;
    for (int i = t; i < n; i += 256) s += row[i];
    red[t] = s; __syncthreads();
    for (int o = 128; o > 0; o >>= 1) { if (t < o) red[t] += red[t + o]; __syncthreads(); }
    float mean = red[0] / (float)n; __syncthreads();

    float vs = 0.0f;
    for (int i = t; i < n; i += 256) { float d = row[i] - mean; vs += d * d; }
    red[t] = vs; __syncthreads();
    for (int o = 128; o > 0; o >>= 1) { if (t < o) red[t] += red[t + o]; __syncthreads(); }
    float rstd = rsqrtf(red[0] / (float)n + 1e-5f); __syncthreads();

    for (int i = t; i < n; i += 256)
        row[i] = apply_act((row[i] - mean) * rstd * g[i] + b[i], act);
}

// ---------------------------------------------------------------------------
// y[N] = act( x[K] @ W (+bias) ).  transW=1: y[n] = sum_k x[k]*W[n*K+k]
// ---------------------------------------------------------------------------
__global__ void vecmat_kernel(const float* __restrict__ x, const float* __restrict__ W,
                              int transW, const float* __restrict__ bias,
                              float* __restrict__ y, int N, int K, int act)
{
    int n = blockIdx.x * blockDim.x + threadIdx.x;
    if (n >= N) return;
    float s = bias ? bias[n] : 0.0f;
    if (transW) { for (int k = 0; k < K; ++k) s += x[k] * W[(size_t)n * K + k]; }
    else        { for (int k = 0; k < K; ++k) s += x[k] * W[(size_t)k * N + n]; }
    y[n] = apply_act(s, act);
}

// agent_state[a][e] = mean_n feats_s[a][n][e] + internal[a][e]
__global__ void agent_state_kernel(const float* __restrict__ feats_s,
                                   const float* __restrict__ internal,
                                   float* __restrict__ out)
{
    int a = blockIdx.x, e = threadIdx.x;
    const float* base = feats_s + ((size_t)a * NSUP) * EDIM + e;
    float s = 0.0f;
    for (int n = 0; n < NSUP; ++n) s += base[(size_t)n * EDIM];
    out[a * EDIM + e] = s / (float)NSUP + internal[a * EDIM + e];
}

// ---------------------------------------------------------------------------
// Communication round: per-agent 4-token MHA(h=4) aggregation + process MLP.
// One block per agent, 256 threads.
// ---------------------------------------------------------------------------
__global__ __launch_bounds__(256)
void comm_round_kernel(const float* __restrict__ messages, const float* __restrict__ comm_w,
                       const float* __restrict__ agg_in_w, const float* __restrict__ agg_in_b,
                       const float* __restrict__ agg_out_w, const float* __restrict__ agg_out_b,
                       const float* __restrict__ internal,
                       const float* __restrict__ proc_w1, const float* __restrict__ proc_b1,
                       const float* __restrict__ proc_w2, const float* __restrict__ proc_b2,
                       float* __restrict__ ck)
{
    __shared__ float tok[4][EDIM];
    __shared__ float qkv[4][3 * EDIM];
    __shared__ float prob[4][4][4];    // [head][i][j]
    __shared__ float ov[4][EDIM];
    __shared__ float comb[EDIM];
    __shared__ float hh[EDIM];

    int a = blockIdx.x, t = threadIdx.x;

    // softmax(comm_w[a,:])  (A=5, recompute per thread)
    float w[AGENTS]; float mx = -1e30f;
    for (int j = 0; j < AGENTS; ++j) { w[j] = comm_w[a * AGENTS + j]; mx = fmaxf(mx, w[j]); }
    float sw = 0.0f;
    for (int j = 0; j < AGENTS; ++j) { w[j] = expf(w[j] - mx); sw += w[j]; }

    int idx = 0;
    for (int j = 0; j < AGENTS; ++j) {
        if (j == a) continue;
        tok[idx][t] = messages[j * EDIM + t] * (w[j] / sw);
        ++idx;
    }
    __syncthreads();

    // qkv = tok @ agg_in_w[a].T + agg_in_b[a]   (4 x 768)
    for (int ti = 0; ti < 4; ++ti)
        for (int c = 0; c < 3; ++c) {
            int r = c * 256 + t;
            const float* Wr = agg_in_w + ((size_t)a * 768 + r) * EDIM;
            float s = agg_in_b[a * 768 + r];
            for (int e = 0; e < EDIM; ++e) s += tok[ti][e] * Wr[e];
            qkv[ti][r] = s;
        }
    __syncthreads();

    if (t < 64) {   // scores, 4 heads x 4 x 4, hd=64, scale 1/8
        int h = t >> 4, i = (t >> 2) & 3, j = t & 3;
        float s = 0.0f;
        for (int d = 0; d < 64; ++d) s += qkv[i][h * 64 + d] * qkv[j][256 + h * 64 + d];
        prob[h][i][j] = s * 0.125f;
    }
    __syncthreads();
    if (t < 16) {   // softmax over j
        int h = t >> 2, i = t & 3;
        float m2 = -1e30f;
        for (int j = 0; j < 4; ++j) m2 = fmaxf(m2, prob[h][i][j]);
        float z = 0.0f;
        for (int j = 0; j < 4; ++j) { prob[h][i][j] = expf(prob[h][i][j] - m2); z += prob[h][i][j]; }
        for (int j = 0; j < 4; ++j) prob[h][i][j] /= z;
    }
    __syncthreads();

    {   // o = P @ V  (per e = h*64+d)
        int h = t >> 6;
        for (int i = 0; i < 4; ++i) {
            float s = 0.0f;
            for (int j = 0; j < 4; ++j) s += prob[h][i][j] * qkv[j][512 + t];
            ov[i][t] = s;
        }
    }
    __syncthreads();

    {   // proj (o @ out_w.T + out_b), mean over tokens, + internal
        const float* We = agg_out_w + ((size_t)a * EDIM + t) * EDIM;
        float s = 0.0f;
        for (int i = 0; i < 4; ++i) {
            float pi = 0.0f;
            for (int k = 0; k < EDIM; ++k) pi += ov[i][k] * We[k];
            s += pi;
        }
        comb[t] = internal[a * EDIM + t] + s * 0.25f + agg_out_b[a * EDIM + t];
    }
    __syncthreads();

    {   // hh = relu(comb @ proc_w1[a] + b1)
        float s = proc_b1[a * EDIM + t];
        for (int e = 0; e < EDIM; ++e) s += comb[e] * proc_w1[((size_t)a * EDIM + e) * EDIM + t];
        hh[t] = fmaxf(s, 0.0f);
    }
    __syncthreads();

    {   // ck = hh @ proc_w2[a] + b2
        float s = proc_b2[a * EDIM + t];
        for (int e = 0; e < EDIM; ++e) s += hh[e] * proc_w2[((size_t)a * EDIM + e) * EDIM + t];
        ck[a * EDIM + t] = s;
    }
}

// Masked row softmax over scores[S,S]; mask_jk = (y[j]==y[row]); in place.
__global__ __launch_bounds__(256)
void softmax_mask_kernel(float* __restrict__ s, const int* __restrict__ y, int S)
{
    __shared__ float red[256];
    int r = blockIdx.x, t = threadIdx.x;
    float* row = s + (size_t)r * S;
    int yr = y[r];

    float mx = -1e30f;
    for (int j = t; j < S; j += 256) {
        float v = (y[j] == yr) ? row[j] : -1e9f;
        mx = fmaxf(mx, v);
    }
    red[t] = mx; __syncthreads();
    for (int o = 128; o > 0; o >>= 1) { if (t < o) red[t] = fmaxf(red[t], red[t + o]); __syncthreads(); }
    float M = red[0]; __syncthreads();

    float sum = 0.0f;
    for (int j = t; j < S; j += 256) {
        float v = (y[j] == yr) ? row[j] : -1e9f;
        sum += expf(v - M);
    }
    red[t] = sum; __syncthreads();
    for (int o = 128; o > 0; o >>= 1) { if (t < o) red[t] += red[t + o]; __syncthreads(); }
    float Z = red[0]; __syncthreads();

    for (int j = t; j < S; j += 256) {
        float v = (y[j] == yr) ? row[j] : -1e9f;
        row[j] = expf(v - M) / Z;
    }
}

// protos[c][e] = mean over {n: y[n]==c} of att[n][e] (0 if empty). Block per class.
__global__ void protos_kernel(const float* __restrict__ att, const int* __restrict__ y,
                              float* __restrict__ protos)
{
    int c = blockIdx.x, e = threadIdx.x;
    float s = 0.0f; int cnt = 0;
    for (int n = 0; n < NSUP; ++n)
        if (y[n] == c) { s += att[(size_t)n * EDIM + e]; ++cnt; }
    protos[c * EDIM + e] = (cnt > 0) ? s / (float)cnt : 0.0f;
}

// out[m] = X[m,:] . w + *bias
__global__ void rowdot_kernel(const float* __restrict__ X, const float* __restrict__ w,
                              const float* __restrict__ bias, float* __restrict__ out,
                              int M, int K)
{
    int m = blockIdx.x * blockDim.x + threadIdx.x;
    if (m >= M) return;
    float s = bias[0];
    const float* row = X + (size_t)m * K;
    for (int k = 0; k < K; ++k) s += row[k] * w[k];
    out[m] = s;
}

// logits[q][c] = -sqrt(max(|fq|^2+|p_c|^2-2 fq.p_c, 1e-12)) + 0.1*conf[q]
__global__ void logits_kernel(const float* __restrict__ fq, const float* __restrict__ protos,
                              const float* __restrict__ conf, float* __restrict__ out)
{
    int q = blockIdx.x * 256 + threadIdx.x;
    if (q >= NQRY) return;
    const float* f = fq + (size_t)q * EDIM;
    float fn = 0.0f;
    for (int e = 0; e < EDIM; ++e) fn += f[e] * f[e];
    float cf = conf[q] * 0.1f;
    for (int c = 0; c < NWAYC; ++c) {
        const float* p = protos + c * EDIM;
        float pn = 0.0f, d = 0.0f;
        for (int e = 0; e < EDIM; ++e) { pn += p[e] * p[e]; d += f[e] * p[e]; }
        float d2 = fn + pn - 2.0f * d;
        out[(size_t)q * NWAYC + c] = -sqrtf(fmaxf(d2, 1e-12f)) + cf;
    }
}

// qcat[q][a*E+e] = feats_q[a][q][e]
__global__ void qcat_kernel(const float* __restrict__ fq, float* __restrict__ qcat)
{
    size_t i = (size_t)blockIdx.x * 256 + threadIdx.x;
    if (i >= (size_t)NQRY * AGENTS * EDIM) return;
    size_t q = i / (AGENTS * EDIM);
    int rem = (int)(i % (AGENTS * EDIM));
    int a = rem / EDIM, e = rem % EDIM;
    qcat[i] = fq[((size_t)a * NQRY + q) * EDIM + e];
}

// ---------------------------------------------------------------------------
static inline void gemm(hipStream_t st, const float* A, int lda,
                        const float* B, int ldb, int transB,
                        float* C, int ldc, int M, int N, int K,
                        const float* bias, float alpha, int act)
{
    dim3 grid(N / TILE_N, M / TILE_M);
    wmma_gemm_kernel<<<grid, 256, 0, st>>>(A, lda, B, ldb, transB, C, ldc,
                                           M, N, K, bias, alpha, act);
}

extern "C" void kernel_launch(void* const* d_in, const int* in_sizes, int n_in,
                              void* d_out, int out_size, void* d_ws, size_t ws_size,
                              hipStream_t stream)
{
    (void)in_sizes; (void)n_in; (void)out_size; (void)ws_size;
    const float* support_x = (const float*)d_in[0];
    const int*   support_y = (const int*)d_in[1];
    const float* query_x   = (const float*)d_in[2];
    const float* fe0_w1 = (const float*)d_in[4],  *fe0_b1 = (const float*)d_in[5];
    const float* fe0_g1 = (const float*)d_in[6],  *fe0_be1= (const float*)d_in[7];
    const float* fe0_w2 = (const float*)d_in[8],  *fe0_b2 = (const float*)d_in[9];
    const float* fe0_g2 = (const float*)d_in[10], *fe0_be2= (const float*)d_in[11];
    const float* fe1_w1 = (const float*)d_in[12], *fe1_b1 = (const float*)d_in[13];
    const float* fe1_g  = (const float*)d_in[14], *fe1_be = (const float*)d_in[15];
    const float* fe1_w2 = (const float*)d_in[16], *fe1_b2 = (const float*)d_in[17];
    const float* fe2_w1 = (const float*)d_in[18], *fe2_b1 = (const float*)d_in[19];
    const float* fe2_w2 = (const float*)d_in[20], *fe2_b2 = (const float*)d_in[21];
    const float* fe2_g  = (const float*)d_in[22], *fe2_be = (const float*)d_in[23];
    const float* fe3_w1 = (const float*)d_in[24], *fe3_b1 = (const float*)d_in[25];
    const float* fe3_w2 = (const float*)d_in[26], *fe3_b2 = (const float*)d_in[27];
    const float* fe3_g  = (const float*)d_in[28], *fe3_be = (const float*)d_in[29];
    const float* fe4_w1 = (const float*)d_in[30], *fe4_b1 = (const float*)d_in[31];
    const float* fe4_w2 = (const float*)d_in[32], *fe4_b2 = (const float*)d_in[33];
    const float* fe4_w3 = (const float*)d_in[34], *fe4_b3 = (const float*)d_in[35];
    const float* msg_enc_w = (const float*)d_in[36], *msg_enc_b = (const float*)d_in[37];
    const float* agg_in_w  = (const float*)d_in[38], *agg_in_b  = (const float*)d_in[39];
    const float* agg_out_w = (const float*)d_in[40], *agg_out_b = (const float*)d_in[41];
    const float* proc_w1 = (const float*)d_in[42], *proc_b1 = (const float*)d_in[43];
    const float* proc_w2 = (const float*)d_in[44], *proc_b2 = (const float*)d_in[45];
    const float* internal_state = (const float*)d_in[46];
    const float* proto_in_w  = (const float*)d_in[47], *proto_in_b  = (const float*)d_in[48];
    const float* proto_out_w = (const float*)d_in[49], *proto_out_b = (const float*)d_in[50];
    const float* cls_w1 = (const float*)d_in[51], *cls_b1 = (const float*)d_in[52];
    const float* cls_w2 = (const float*)d_in[53], *cls_b2 = (const float*)d_in[54];
    const float* cls_w3 = (const float*)d_in[55], *cls_b3 = (const float*)d_in[56];
    const float* comm_w = (const float*)d_in[57];
    const float* f_w1 = (const float*)d_in[58], *f_b1 = (const float*)d_in[59];
    const float* f_w2 = (const float*)d_in[60], *f_b2 = (const float*)d_in[61];
    const float* f_w3 = (const float*)d_in[62], *f_b3 = (const float*)d_in[63];

    float* wsf = (float*)d_ws;
    float* out = (float*)d_out;

    // workspace layout (floats)
    size_t off = 0;
    auto alloc = [&](size_t n) { size_t o = off; off += n; return o; };
    const size_t featS_o  = alloc((size_t)AGENTS * NSUP * EDIM);
    const size_t featQ_o  = alloc((size_t)AGENTS * NQRY * EDIM);
    const size_t tmpA_o   = alloc((size_t)NQRY * 768);      // FE temp; reused as qcat
    const size_t tmpB_o   = alloc((size_t)NQRY * 768);      // FE temp (qcat spills here)
    const size_t astate_o = alloc(AGENTS * EDIM);
    const size_t msgs_o   = alloc(AGENTS * EDIM);
    const size_t ck_o     = alloc(AGENTS * EDIM);
    const size_t bias2_o  = alloc(3 * EDIM);
    const size_t biasC_o  = alloc(EDIM);
    const size_t qkv_o    = alloc((size_t)NSUP * 3 * EDIM);
    const size_t sc_o     = alloc(2 * (size_t)NSUP * NSUP);
    const size_t obuf_o   = alloc((size_t)NSUP * EDIM);
    const size_t att_o    = alloc((size_t)NSUP * EDIM);
    const size_t protos_o = alloc(AGENTS * NWAYC * EDIM);
    const size_t conf_o   = alloc((size_t)AGENTS * NQRY);
    const size_t h1_o     = alloc((size_t)NQRY * EDIM);     // reused as fusion h1
    const size_t h2_o     = alloc((size_t)NQRY * (EDIM / 2)); // reused as fusion h2

    float* featS = wsf + featS_o;
    float* featQ = wsf + featQ_o;
    float* t1 = wsf + tmpA_o;
    float* t2 = wsf + tmpB_o;

    // ---- Stage 1: per-agent feature extractors (WMMA GEMM chains) ----
    auto run_fes = [&](const float* X, int M, float* feats) {
        float* f;
        // fe0: LN(relu) -> LN(relu)
        gemm(stream, X, DDIM, fe0_w1, 512, 0, t1, 512, M, 512, DDIM, fe0_b1, 1.f, 0);
        ln_act_kernel<<<M, 256, 0, stream>>>(t1, 512, fe0_g1, fe0_be1, 1);
        f = feats + 0 * (size_t)M * EDIM;
        gemm(stream, t1, 512, fe0_w2, 256, 0, f, 256, M, 256, 512, fe0_b2, 1.f, 0);
        ln_act_kernel<<<M, 256, 0, stream>>>(f, 256, fe0_g2, fe0_be2, 1);
        // fe1: LN(relu) -> tanh
        gemm(stream, X, DDIM, fe1_w1, 256, 0, t1, 256, M, 256, DDIM, fe1_b1, 1.f, 0);
        ln_act_kernel<<<M, 256, 0, stream>>>(t1, 256, fe1_g, fe1_be, 1);
        f = feats + 1 * (size_t)M * EDIM;
        gemm(stream, t1, 256, fe1_w2, 256, 0, f, 256, M, 256, 256, fe1_b2, 1.f, 2);
        // fe2: relu -> LN(relu)
        gemm(stream, X, DDIM, fe2_w1, 768, 0, t1, 768, M, 768, DDIM, fe2_b1, 1.f, 1);
        f = feats + 2 * (size_t)M * EDIM;
        gemm(stream, t1, 768, fe2_w2, 256, 0, f, 256, M, 256, 768, fe2_b2, 1.f, 0);
        ln_act_kernel<<<M, 256, 0, stream>>>(f, 256, fe2_g, fe2_be, 1);
        // fe3: relu -> LN(sigmoid)
        gemm(stream, X, DDIM, fe3_w1, 256, 0, t1, 256, M, 256, DDIM, fe3_b1, 1.f, 1);
        f = feats + 3 * (size_t)M * EDIM;
        gemm(stream, t1, 256, fe3_w2, 256, 0, f, 256, M, 256, 256, fe3_b2, 1.f, 0);
        ln_act_kernel<<<M, 256, 0, stream>>>(f, 256, fe3_g, fe3_be, 3);
        // fe4: relu -> relu -> linear
        gemm(stream, X, DDIM, fe4_w1, 512, 0, t1, 512, M, 512, DDIM, fe4_b1, 1.f, 1);
        gemm(stream, t1, 512, fe4_w2, 256, 0, t2, 256, M, 256, 512, fe4_b2, 1.f, 1);
        f = feats + 4 * (size_t)M * EDIM;
        gemm(stream, t2, 256, fe4_w3, 256, 0, f, 256, M, 256, 256, fe4_b3, 1.f, 0);
    };
    run_fes(support_x, NSUP, featS);
    run_fes(query_x, NQRY, featQ);

    // ---- Stage 2: communication round -> ck[A,E] ----
    agent_state_kernel<<<AGENTS, 256, 0, stream>>>(featS, internal_state, wsf + astate_o);
    for (int a = 0; a < AGENTS; ++a)
        vecmat_kernel<<<1, 256, 0, stream>>>(wsf + astate_o + a * EDIM,
                                             msg_enc_w + (size_t)a * EDIM * EDIM, 0,
                                             msg_enc_b + a * EDIM,
                                             wsf + msgs_o + a * EDIM, EDIM, EDIM, 2);
    comm_round_kernel<<<AGENTS, 256, 0, stream>>>(wsf + msgs_o, comm_w,
                                                  agg_in_w, agg_in_b, agg_out_w, agg_out_b,
                                                  internal_state, proc_w1, proc_b1,
                                                  proc_w2, proc_b2, wsf + ck_o);

    // ---- Stage 3: per-agent masked prototype attention + classifier ----
    const int* y = support_y;
    for (int a = 0; a < AGENTS; ++a) {
        const float* fs = featS + (size_t)a * NSUP * EDIM;
        const float* fq = featQ + (size_t)a * NQRY * EDIM;
        float* qkvp = wsf + qkv_o;

        // qkv = (fs + ck) @ in_w.T + in_b   (ck folded into bias)
        vecmat_kernel<<<3, 256, 0, stream>>>(wsf + ck_o + a * EDIM,
                                             proto_in_w + (size_t)a * 768 * EDIM, 1,
                                             proto_in_b + a * 768, wsf + bias2_o, 768, EDIM, 0);
        gemm(stream, fs, EDIM, proto_in_w + (size_t)a * 768 * EDIM, EDIM, 1,
             qkvp, 768, NSUP, 768, EDIM, wsf + bias2_o, 1.f, 0);

        for (int h = 0; h < 2; ++h) {   // 2 heads, hd=128
            float* sc = wsf + sc_o + (size_t)h * NSUP * NSUP;
            gemm(stream, qkvp + h * 128, 768, qkvp + 256 + h * 128, 768, 1,
                 sc, NSUP, NSUP, NSUP, 128, nullptr, 0.08838834764831845f, 0);
            softmax_mask_kernel<<<NSUP, 256, 0, stream>>>(sc, y, NSUP);
            gemm(stream, sc, NSUP, qkvp + 512 + h * 128, 768, 0,
                 wsf + obuf_o + h * 128, EDIM, NSUP, 128, NSUP, nullptr, 1.f, 0);
        }
        gemm(stream, wsf + obuf_o, EDIM, proto_out_w + (size_t)a * EDIM * EDIM, EDIM, 1,
             wsf + att_o, EDIM, NSUP, EDIM, EDIM, proto_out_b + a * EDIM, 1.f, 0);
        protos_kernel<<<NWAYC, 256, 0, stream>>>(wsf + att_o, y,
                                                 wsf + protos_o + a * NWAYC * EDIM);

        // conf MLP: eq = [fq, ck] -> relu -> relu -> scalar ; ck half folded to bias
        vecmat_kernel<<<1, 256, 0, stream>>>(wsf + ck_o + a * EDIM,
                                             cls_w1 + (size_t)a * 512 * EDIM + 256 * EDIM, 0,
                                             cls_b1 + a * EDIM, wsf + biasC_o, EDIM, EDIM, 0);
        gemm(stream, fq, EDIM, cls_w1 + (size_t)a * 512 * EDIM, EDIM, 0,
             wsf + h1_o, EDIM, NQRY, EDIM, EDIM, wsf + biasC_o, 1.f, 1);
        gemm(stream, wsf + h1_o, EDIM, cls_w2 + (size_t)a * EDIM * 128, 128, 0,
             wsf + h2_o, 128, NQRY, 128, EDIM, cls_b2 + a * 128, 1.f, 1);
        rowdot_kernel<<<NQRY / 256, 256, 0, stream>>>(wsf + h2_o, cls_w3 + a * 128,
                                                      cls_b3 + a, wsf + conf_o + (size_t)a * NQRY,
                                                      NQRY, 128);
        logits_kernel<<<NQRY / 256, 256, 0, stream>>>(fq, wsf + protos_o + a * NWAYC * EDIM,
                                                      wsf + conf_o + (size_t)a * NQRY,
                                                      out + (size_t)a * NQRY * NWAYC);
    }

    // ---- Stage 4: decision fusion (qcat reuses FE temp region) ----
    float* qcatp = wsf + tmpA_o;   // NQ*1280 fits in tmpA+tmpB (2 * NQ*768)
    {
        size_t total = (size_t)NQRY * AGENTS * EDIM;
        qcat_kernel<<<(unsigned)((total + 255) / 256), 256, 0, stream>>>(featQ, qcatp);
    }
    gemm(stream, qcatp, AGENTS * EDIM, f_w1, 256, 0, wsf + h1_o, 256,
         NQRY, 256, AGENTS * EDIM, f_b1, 1.f, 1);
    gemm(stream, wsf + h1_o, 256, f_w2, 128, 0, wsf + h2_o, 128,
         NQRY, 128, 256, f_b2, 1.f, 1);
    rowdot_kernel<<<NQRY / 256, 256, 0, stream>>>(wsf + h2_o, f_w3, f_b3,
                                                  out + (size_t)AGENTS * NQRY * NWAYC,
                                                  NQRY, 128);
}
